// StatementEncoder_37623913513181
// MI455X (gfx1250) — matrix operand order repaired and checked by reading, not compiled, not run on hardware
//
#include <hip/hip_runtime.h>
#include <hip/hip_bf16.h>
#include <stdint.h>

// ---------------------------------------------------------------------------
// Types for CDNA5 WMMA (wave32, v_wmma_f32_16x16x32_bf16)
// ---------------------------------------------------------------------------
typedef __attribute__((ext_vector_type(16))) __bf16 v16bf;
typedef __attribute__((ext_vector_type(8)))  float  v8f;

__device__ __forceinline__ uint32_t f2bf_pk(float lo, float hi) {
  uint32_t a = __float_as_uint(lo), b = __float_as_uint(hi);
  a = (a + 0x7FFFu + ((a >> 16) & 1u)) >> 16;   // RNE fp32 -> bf16
  b = (b + 0x7FFFu + ((b >> 16) & 1u)) >> 16;
  return (a & 0xFFFFu) | ((b & 0xFFFFu) << 16);
}

__device__ __forceinline__ void atomAddF(float* p, float v) {
  __hip_atomic_fetch_add(p, v, __ATOMIC_RELAXED, __HIP_MEMORY_SCOPE_AGENT);
}

__device__ __forceinline__ void atomicMaxF(float* addr, float v) {
  int cur = __float_as_int(*addr);
  while (v > __int_as_float(cur)) {
    int prev = atomicCAS((int*)addr, cur, __float_as_int(v));
    if (prev == cur) break;
    cur = prev;
  }
}

// CDNA5 async global->LDS copy (ASYNCcnt-tracked, per-lane 16B)
__device__ __forceinline__ void async_copy16(uint32_t lds_off, const void* gsrc) {
  asm volatile("global_load_async_to_lds_b128 %0, %1, off"
               :: "v"(lds_off), "v"(gsrc)
               : "memory");
}
__device__ __forceinline__ void wait_asynccnt0() {
  asm volatile("s_wait_asynccnt 0" ::: "memory");
}

// ---------------------------------------------------------------------------
// WMMA GEMM: C[M,Nd] = act(A[M,Kd] @ W[Kd,Nd] + bias)
// Block: 256 threads = 8 waves; block tile = 128 rows x 64 cols.
// Each wave owns 16 rows and 4 accumulators (4 n-tiles); per 32-K chunk it
// issues 4 WMMAs off one shared A fragment.
// A tile staged raw fp32 via global_load_async_to_lds_b128; converted to bf16
// at fragment-read time. B tile staged transposed+converted synchronously.
// flags: bit0 = add bias, bit1 = relu
// ---------------------------------------------------------------------------
__global__ __launch_bounds__(256) void gemm_wmma_kernel(
    const float* __restrict__ A, const float* __restrict__ W,
    const float* __restrict__ bias, float* __restrict__ C,
    int M, int Kd, int Nd, int flags)
{
  __shared__ __align__(16) float    Af[128][32];  // raw fp32 A tile (16 KB)
  __shared__ __align__(16) uint32_t Bs[64][16];   // 64 n x 32 k bf16-pairs (4 KB)

  const int tid  = threadIdx.x;
  const int lane = tid & 31, wv = tid >> 5;
  const long long m0 = (long long)blockIdx.x * 128;
  const int n0 = blockIdx.y * 64;

  const int rA = tid >> 1, halfA = tid & 1;       // A staging: 2 thr/row, 16 floats each

  v8f acc[4] = {v8f{}, v8f{}, v8f{}, v8f{}};

  const int nchunks = Kd >> 5;
  for (int kb = 0; kb < nchunks; ++kb) {
    // ---- stage A tile via async DMA (128 x 32 fp32) ----
    {
      long long gr = m0 + rA;
      uint32_t lds_off = (uint32_t)(uintptr_t)&Af[rA][halfA * 16];
      if (gr < M) {
        const float* src = A + gr * (long long)Kd + kb * 32 + halfA * 16;
#pragma unroll
        for (int i = 0; i < 4; ++i)
          async_copy16(lds_off + i * 16, src + i * 4);
      } else {
        float4 z = {0.f, 0.f, 0.f, 0.f};
#pragma unroll
        for (int i = 0; i < 4; ++i)
          *(float4*)&Af[rA][halfA * 16 + 4 * i] = z;
      }
    }
    // ---- stage B tile transposed: Bs[n][k/2] over 64 n ----
    {
#pragma unroll
      for (int j = 0; j < 4; ++j) {
        int id = tid + 256 * j;
        int nn = id & 63, cc = id >> 6;
        long long kk = (long long)kb * 32 + 2 * cc;
        Bs[nn][cc] = f2bf_pk(W[kk * Nd + n0 + nn], W[(kk + 1) * Nd + n0 + nn]);
      }
    }
    wait_asynccnt0();
    __syncthreads();

    // ---- A fragment (fp32 -> bf16 at read) per ISA 7.12.2 layout ----
    union { uint32_t u[8]; v16bf v; } fa;
    const int mloc = lane & 15, hh = lane >> 4;
    {
      const float* arow = &Af[wv * 16 + mloc][0];
      float4 c0 = *(const float4*)(arow + hh * 8);           // k = hh*8 .. +3
      float4 c1 = *(const float4*)(arow + hh * 8 + 4);       // k = hh*8+4 .. +7
      float4 c2 = *(const float4*)(arow + 16 + hh * 8);      // k = 16+hh*8 ..
      float4 c3 = *(const float4*)(arow + 16 + hh * 8 + 4);
      fa.u[0] = f2bf_pk(c0.x, c0.y); fa.u[1] = f2bf_pk(c0.z, c0.w);
      fa.u[2] = f2bf_pk(c1.x, c1.y); fa.u[3] = f2bf_pk(c1.z, c1.w);
      fa.u[4] = f2bf_pk(c2.x, c2.y); fa.u[5] = f2bf_pk(c2.z, c2.w);
      fa.u[6] = f2bf_pk(c3.x, c3.y); fa.u[7] = f2bf_pk(c3.z, c3.w);
    }
    // ---- 4 n-tiles: B fragment + WMMA each ----
#pragma unroll
    for (int t = 0; t < 4; ++t) {
      union { uint32_t u[8]; v16bf v; } fb;
      const uint32_t* brow = &Bs[t * 16 + mloc][0];
      uint4 s0 = *(const uint4*)(brow + hh * 4);             // k = hh*8 .. +7
      uint4 s1 = *(const uint4*)(brow + 8 + hh * 4);
      fb.u[0] = s0.x; fb.u[1] = s0.y; fb.u[2] = s0.z; fb.u[3] = s0.w;
      fb.u[4] = s1.x; fb.u[5] = s1.y; fb.u[6] = s1.z; fb.u[7] = s1.w;
      acc[t] = __builtin_amdgcn_wmma_f32_16x16x32_bf16(
          false, fa.v, false, fb.v, (short)0, acc[t], false, false);
    }
    __syncthreads();
  }

  // ---- epilogue: C/D layout lane=N, dword v -> M = v + (lane/16)*8 ----
  const int nloc = lane & 15, hh2 = lane >> 4;
#pragma unroll
  for (int t = 0; t < 4; ++t) {
    int nc = n0 + t * 16 + nloc;
#pragma unroll
    for (int v = 0; v < 8; ++v) {
      long long gm = m0 + wv * 16 + hh2 * 8 + v;
      if (gm < M) {
        float val = acc[t][v];
        if (flags & 1) val += bias[nc];
        if (flags & 2) val = fmaxf(val, 0.0f);
        C[gm * (long long)Nd + nc] = val;
      }
    }
  }
}

// ---------------------------------------------------------------------------
// Utility / graph kernels
// ---------------------------------------------------------------------------
__global__ void fill_u32_kernel(uint32_t* p, uint32_t pat, long long n) {
  long long i = (long long)blockIdx.x * blockDim.x + threadIdx.x;
  long long stride = (long long)gridDim.x * blockDim.x;
  for (; i < n; i += stride) p[i] = pat;
}

// per-(node,head) attention logits: one wave per (node,head)
__global__ void alpha_kernel(const float* __restrict__ h,
                             const float* __restrict__ a_s, const float* __restrict__ a_d,
                             float* asv, float* adv, int N, int H) {
  long long wid = ((long long)blockIdx.x * blockDim.x + threadIdx.x) >> 5;
  int lane = threadIdx.x & 31;
  if (wid >= (long long)N * H) return;
  int hd = (int)(wid % H);
  const float* hp = h + wid * 128;
  const float* sv = a_s + hd * 128;
  const float* dv = a_d + hd * 128;
  float s = 0.f, d = 0.f;
  for (int f = lane; f < 128; f += 32) { float v = hp[f]; s += v * sv[f]; d += v * dv[f]; }
  for (int off = 16; off; off >>= 1) { s += __shfl_xor(s, off); d += __shfl_xor(d, off); }
  if (lane == 0) { asv[wid] = s; adv[wid] = d; }
}

__device__ __forceinline__ void edge_sd(const int* ei, int E, long long eid, int& s, int& d) {
  if (eid < E) { s = ei[eid]; d = ei[(long long)E + eid]; }
  else         { s = d = (int)(eid - E); }   // self loops appended
}

__global__ void edge_alpha_max(const int* __restrict__ ei, const float* __restrict__ asv,
                               const float* __restrict__ adv, float* ee, float* mmax,
                               int E, int N, int H) {
  long long idx = (long long)blockIdx.x * blockDim.x + threadIdx.x;
  long long total = ((long long)E + N) * H;
  if (idx >= total) return;
  long long eid = idx / H; int hd = (int)(idx % H);
  int s, d; edge_sd(ei, E, eid, s, d);
  float a = asv[(long long)s * H + hd] + adv[(long long)d * H + hd];
  a = a > 0.f ? a : 0.2f * a;               // leaky_relu(0.2)
  ee[idx] = a;
  atomicMaxF(&mmax[(long long)d * H + hd], a);
}

__global__ void edge_exp_sum(const int* __restrict__ ei, float* ee,
                             const float* __restrict__ mmax, float* ssum,
                             int E, int N, int H) {
  long long idx = (long long)blockIdx.x * blockDim.x + threadIdx.x;
  long long total = ((long long)E + N) * H;
  if (idx >= total) return;
  long long eid = idx / H; int hd = (int)(idx % H);
  int s, d; edge_sd(ei, E, eid, s, d);
  float e = __expf(ee[idx] - mmax[(long long)d * H + hd]);
  ee[idx] = e;
  atomAddF(&ssum[(long long)d * H + hd], e);
}

// one wave per edge; scatter h[src]*coef into out[dst]
__global__ void agg_kernel(const int* __restrict__ ei, const float* __restrict__ ee,
                           const float* __restrict__ ssum, const float* __restrict__ hpre,
                           float* out, int E, int N, int H) {
  long long wid = ((long long)blockIdx.x * blockDim.x + threadIdx.x) >> 5;
  int lane = threadIdx.x & 31;
  long long Etot = (long long)E + N;
  if (wid >= Etot) return;
  int s, d; edge_sd(ei, E, wid, s, d);
  for (int hd = 0; hd < H; ++hd) {
    float coef = ee[wid * H + hd] / (ssum[(long long)d * H + hd] + 1e-16f);
    const float* hs = hpre + ((long long)s * H + hd) * 128;
    float* od = out + ((long long)d * H + hd) * 128;
    for (int f = lane; f < 128; f += 32) atomAddF(&od[f], hs[f] * coef);
  }
}

__global__ void bias_relu_kernel(float* X, const float* __restrict__ b, int C, long long n) {
  long long i = (long long)blockIdx.x * blockDim.x + threadIdx.x;
  if (i >= n) return;
  int c = (int)(i % C);
  X[i] = fmaxf(X[i] + b[c], 0.0f);
}

__global__ void bn_stats_kernel(const float* __restrict__ X, float* sums, float* sumsq,
                                int rows, int C) {
  long long r0 = (long long)blockIdx.x * 256;
  int rend = rows - (int)r0; if (rend > 256) rend = 256; if (rend <= 0) return;
  for (int c = threadIdx.x; c < C; c += blockDim.x) {
    float s = 0.f, q = 0.f;
    for (int r = 0; r < rend; ++r) {
      float v = X[(r0 + r) * (long long)C + c];
      s += v; q += v * v;
    }
    atomAddF(&sums[c], s);
    atomAddF(&sumsq[c], q);
  }
}

__global__ void bn_norm_kernel(float* X, const float* __restrict__ sums,
                               const float* __restrict__ sumsq,
                               const float* __restrict__ g, const float* __restrict__ b,
                               long long rows, int C, int relu) {
  long long n = rows * C;
  long long i = (long long)blockIdx.x * blockDim.x + threadIdx.x;
  if (i >= n) return;
  int c = (int)(i % C);
  float invR = 1.0f / (float)rows;
  float mu  = sums[c] * invR;
  float var = sumsq[c] * invR - mu * mu;
  float v = (X[i] - mu) * rsqrtf(var + 1e-5f) * g[c] + b[c];
  if (relu) v = fmaxf(v, 0.f);
  X[i] = v;
}

__global__ void pnorm_kernel(const float* __restrict__ p, float* invn) {
  __shared__ float red[128];
  int t = threadIdx.x;
  float v = p[t];
  red[t] = v * v; __syncthreads();
  for (int s = 64; s > 0; s >>= 1) { if (t < s) red[t] += red[t + s]; __syncthreads(); }
  if (t == 0) invn[0] = 1.0f / (sqrtf(red[0]) + 1e-16f);
}

__global__ void score_kernel(const float* __restrict__ h, const float* __restrict__ p,
                             const float* __restrict__ invn, float* scores, int N) {
  long long wid = ((long long)blockIdx.x * blockDim.x + threadIdx.x) >> 5;
  int lane = threadIdx.x & 31;
  if (wid >= N) return;
  const float* hp = h + wid * 128;
  float s = 0.f;
  for (int f = lane; f < 128; f += 32) s += hp[f] * p[f];
  for (int off = 16; off; off >>= 1) s += __shfl_xor(s, off);
  if (lane == 0) scores[wid] = tanhf(s * invn[0]);
}

__device__ __forceinline__ int score_bin(float s) {
  int b = (int)((s + 1.0f) * 2048.0f);
  return b < 0 ? 0 : (b > 4095 ? 4095 : b);
}

__global__ void hist_kernel(const float* __restrict__ scores, int* hist, int N) {
  int n = blockIdx.x * blockDim.x + threadIdx.x;
  if (n >= N) return;
  atomicAdd(&hist[score_bin(scores[n])], 1);
}

__global__ void topk_scan_kernel(const int* __restrict__ hist, int K, int* scano) {
  if (threadIdx.x || blockIdx.x) return;
  int cum = 0, T = 0, rem = K;
  for (int b = 4095; b >= 0; --b) {
    int h = hist[b];
    if (cum + h >= K) { T = b; rem = K - cum; break; }
    cum += h;
  }
  scano[0] = T; scano[1] = rem;
}

__global__ void topk_blockcount(const float* __restrict__ scores, int N,
                                const int* __restrict__ scano, int* blkA, int* blkE) {
  __shared__ int cA, cE;
  if (threadIdx.x == 0) { cA = 0; cE = 0; }
  __syncthreads();
  int T = scano[0];
  int n = blockIdx.x * 256 + threadIdx.x;
  if (n < N) {
    int b = score_bin(scores[n]);
    if (b > T) atomicAdd(&cA, 1);
    else if (b == T) atomicAdd(&cE, 1);
  }
  __syncthreads();
  if (threadIdx.x == 0) { blkA[blockIdx.x] = cA; blkE[blockIdx.x] = cE; }
}

__global__ void topk_blockscan(const int* __restrict__ blkA, const int* __restrict__ blkE,
                               int nblk, int* scano, int* aOff, int* eOff, int* eTake) {
  if (threadIdx.x || blockIdx.x) return;
  int rem = scano[1];
  int aAcc = 0, eAcc = 0;
  for (int i = 0; i < nblk; ++i) {
    aOff[i] = aAcc; aAcc += blkA[i];
    int take = blkE[i]; if (take > rem) take = rem;
    eTake[i] = take; eOff[i] = eAcc; eAcc += take; rem -= take;
  }
  scano[2] = aAcc;
}

__global__ void topk_select(const float* __restrict__ scores, int N,
                            const int* __restrict__ scano,
                            const int* __restrict__ aOff, const int* __restrict__ eOff,
                            const int* __restrict__ eTake, int* perm, float* pvals) {
  if (threadIdx.x) return;  // deterministic serial scan per 256-node block
  int T = scano[0], base = scano[2];
  int blk = blockIdx.x;
  int ao = aOff[blk], eo = base + eOff[blk], tk = eTake[blk], taken = 0;
  int n0 = blk * 256, n1 = n0 + 256; if (n1 > N) n1 = N;
  for (int n = n0; n < n1; ++n) {
    float s = scores[n]; int b = score_bin(s);
    if (b > T)                     { perm[ao] = n; pvals[ao] = s; ++ao; }
    else if (b == T && taken < tk) { perm[eo] = n; pvals[eo] = s; ++eo; ++taken; }
  }
}

__global__ void gather_kernel(const float* __restrict__ h, const int* __restrict__ perm,
                              const float* __restrict__ pvals, float* out, int K) {
  long long i = (long long)blockIdx.x * blockDim.x + threadIdx.x;
  if (i >= (long long)K * 128) return;
  int row = (int)(i >> 7), f = (int)(i & 127);
  int n = perm[row];
  float v = h[(long long)n * 128 + f] * pvals[row];
  out[i] = fmaxf(v, 0.0f);
}

// ---------------------------------------------------------------------------
// Host-side orchestration
// ---------------------------------------------------------------------------
static inline unsigned gblk(long long n, int b) { return (unsigned)((n + b - 1) / b); }

extern "C" void kernel_launch(void* const* d_in, const int* in_sizes, int n_in,
                              void* d_out, int out_size, void* d_ws, size_t ws_size,
                              hipStream_t stream) {
  const float* x      = (const float*)d_in[0];
  const int*   ei     = (const int*)d_in[1];
  const float* W0     = (const float*)d_in[2];
  const float* a_src0 = (const float*)d_in[3];
  const float* a_dst0 = (const float*)d_in[4];
  const float* b0     = (const float*)d_in[5];
  const float* bn0_g  = (const float*)d_in[6];
  const float* bn0_b  = (const float*)d_in[7];
  const float* W1     = (const float*)d_in[8];
  const float* a_src1 = (const float*)d_in[9];
  const float* a_dst1 = (const float*)d_in[10];
  const float* b1     = (const float*)d_in[11];
  const float* bn1_g  = (const float*)d_in[12];
  const float* bn1_b  = (const float*)d_in[13];
  const float* p_pool = (const float*)d_in[14];
  const float* mW1    = (const float*)d_in[15];
  const float* mb1    = (const float*)d_in[16];
  const float* mbn_g  = (const float*)d_in[17];
  const float* mbn_b  = (const float*)d_in[18];
  const float* mW2    = (const float*)d_in[19];
  const float* mb2    = (const float*)d_in[20];
  (void)n_in; (void)ws_size;

  const int N  = in_sizes[0] / 128;
  const int E  = in_sizes[1] / 2;
  const int Kp = out_size / 128;             // ENC = 128
  const long long Etot = (long long)E + N;
  const int NBLK = (N + 255) / 256;

  // ---- workspace layout (floats) ----
  float* ws = (float*)d_ws;
  size_t o = 0;
  float* bufA   = ws + o; o += (size_t)N * 384;   // pre-agg h, later pooled rows
  float* bufB   = ws + o; o += (size_t)N * 384;   // aggregated/normalized h, MLP hidden
  float* asv    = ws + o; o += (size_t)N * 3;
  float* adv    = ws + o; o += (size_t)N * 3;
  float* mmax   = ws + o; o += (size_t)N * 3;
  float* ssum   = ws + o; o += (size_t)N * 3;
  float* ee     = ws + o; o += (size_t)Etot * 3;
  float* scores = ws + o; o += (size_t)N;
  float* sums   = ws + o; o += 512;
  float* sumsq  = ws + o; o += 512;
  float* invn   = ws + o; o += 16;
  float* pvals  = ws + o; o += (size_t)Kp;
  int*   perm   = (int*)(ws + o); o += (size_t)Kp;
  int*   hist   = (int*)(ws + o); o += 4096;
  int*   blkA   = (int*)(ws + o); o += (size_t)NBLK;
  int*   blkE   = (int*)(ws + o); o += (size_t)NBLK;
  int*   aOff   = (int*)(ws + o); o += (size_t)NBLK;
  int*   eOff   = (int*)(ws + o); o += (size_t)NBLK;
  int*   eTake  = (int*)(ws + o); o += (size_t)NBLK;
  int*   scano  = (int*)(ws + o); o += 16;

  auto FILL = [&](void* p, uint32_t pat, long long n) {
    unsigned blocks = gblk(n, 256 * 8); if (blocks > 4096) blocks = 4096; if (!blocks) blocks = 1;
    fill_u32_kernel<<<blocks, 256, 0, stream>>>((uint32_t*)p, pat, n);
  };
  auto GEMM = [&](const float* A, const float* W, const float* bias, float* C,
                  long long M, int Kd, int Nd, int flags) {
    dim3 g(gblk(M, 128), (unsigned)(Nd / 64));
    gemm_wmma_kernel<<<g, 256, 0, stream>>>(A, W, bias, C, (int)M, Kd, Nd, flags);
  };
  auto GAT = [&](const float* hpre, const float* a_s, const float* a_d,
                 float* out, int H) {
    alpha_kernel<<<gblk((long long)N * H * 32, 256), 256, 0, stream>>>(
        hpre, a_s, a_d, asv, adv, N, H);
    FILL(mmax, 0xFF7FFFFFu, (long long)N * H);       // -FLT_MAX
    FILL(ssum, 0u, (long long)N * H);
    long long eth = Etot * H;
    edge_alpha_max<<<gblk(eth, 256), 256, 0, stream>>>(ei, asv, adv, ee, mmax, E, N, H);
    edge_exp_sum <<<gblk(eth, 256), 256, 0, stream>>>(ei, ee, mmax, ssum, E, N, H);
    FILL(out, 0u, (long long)N * H * 128);
    agg_kernel<<<gblk(Etot * 32, 256), 256, 0, stream>>>(ei, ee, ssum, hpre, out, E, N, H);
  };
  auto BN = [&](float* X, const float* g, const float* b, long long rows, int C, int relu) {
    FILL(sums, 0u, 512); FILL(sumsq, 0u, 512);
    bn_stats_kernel<<<gblk(rows, 256), 256, 0, stream>>>(X, sums, sumsq, (int)rows, C);
    bn_norm_kernel<<<gblk(rows * C, 256), 256, 0, stream>>>(X, sums, sumsq, g, b, rows, C, relu);
  };

  // ---- layer 0: GATConv(128 -> 3x128) -> ReLU -> BN(384) ----
  GEMM(x, W0, nullptr, bufA, N, 128, 384, 0);                 // h = x @ W0  (WMMA)
  GAT(bufA, a_src0, a_dst0, bufB, 3);                         // segment softmax + scatter
  bias_relu_kernel<<<gblk((long long)N * 384, 256), 256, 0, stream>>>(bufB, b0, 384, (long long)N * 384);
  BN(bufB, bn0_g, bn0_b, N, 384, 0);

  // ---- layer 1: GATConv(384 -> 128) -> ReLU -> BN(128) ----
  GEMM(bufB, W1, nullptr, bufA, N, 384, 128, 0);              // h1 pre-agg (WMMA)
  GAT(bufA, a_src1, a_dst1, bufB, 1);
  bias_relu_kernel<<<gblk((long long)N * 128, 256), 256, 0, stream>>>(bufB, b1, 128, (long long)N * 128);
  BN(bufB, bn1_g, bn1_b, N, 128, 0);

  // ---- TopK pooling (ratio 0.5) ----
  pnorm_kernel<<<1, 128, 0, stream>>>(p_pool, invn);
  score_kernel<<<gblk((long long)N * 32, 256), 256, 0, stream>>>(bufB, p_pool, invn, scores, N);
  FILL(hist, 0u, 4096);
  hist_kernel<<<gblk(N, 256), 256, 0, stream>>>(scores, hist, N);
  topk_scan_kernel<<<1, 32, 0, stream>>>(hist, Kp, scano);
  topk_blockcount<<<NBLK, 256, 0, stream>>>(scores, N, scano, blkA, blkE);
  topk_blockscan<<<1, 32, 0, stream>>>(blkA, blkE, NBLK, scano, aOff, eOff, eTake);
  topk_select<<<NBLK, 32, 0, stream>>>(scores, N, scano, aOff, eOff, eTake, perm, pvals);
  gather_kernel<<<gblk((long long)Kp * 128, 256), 256, 0, stream>>>(bufB, perm, pvals, bufA, Kp);

  // ---- MLP: Linear(128->256) -> BN -> ReLU -> Linear(256->128) ----
  GEMM(bufA, mW1, mb1, bufB, Kp, 128, 256, 1);                // WMMA + bias
  BN(bufB, mbn_g, mbn_b, Kp, 256, 1);                         // BN then ReLU
  GEMM(bufB, mW2, mb2, (float*)d_out, Kp, 256, 128, 1);       // WMMA + bias -> output
}
